// Pointnet2Backbone_52896817217693
// MI455X (gfx1250) — matrix operand order repaired and checked by reading, not compiled, not run on hardware
//
#include <hip/hip_runtime.h>

// ---------------------------------------------------------------------------
// PointNet++ backbone for MI455X (gfx1250, wave32, WMMA).
//  * FPS / ball-query / 3-NN: latency-bound scalar/ballot kernels.
//  * SA stages: fused gather -> 3x MLP (v_wmma_f32_16x16x32_f16, fp32 acc)
//    -> max-pool, activations resident in LDS (up to ~295 KB/WG; 320 KB WGP).
//  * FP stages: fused 3-NN interpolation -> MLP layers, same WMMA engine.
//  * Weights pre-swizzled to f16 fragment layout: B frag = one 32B load.
//  * A frag = two ds_load_b128; K-loop software-pipelined (double buffer)
//    so fragment fetches overlap WMMA execution.
// ---------------------------------------------------------------------------

typedef __attribute__((ext_vector_type(16))) _Float16 v16h;
typedef __attribute__((ext_vector_type(8)))  float    v8f;
typedef __attribute__((ext_vector_type(4)))  unsigned u32x4;

#define WG 256   // 8 wave32 waves per workgroup

union AFrag { v16h v; u32x4 q[2]; };
union CFrag { v8f  v; float f[8]; };

// ---------------------------------------------------------------------------
// Generic LDS->(LDS|pool|global) MLP layer tile engine.
//   X : [rows][cinPad] f16 in LDS (row-major), cinPad % 32 == 0
//   Wp: packed f16 fragments [ktile][ntile][lane][16]
//   mode 0: Y[rows][cout] f16 LDS with bias+relu
//   mode 1: relu + max-pool over 32-row groups into pooledU (f32 bits, >=0)
//   mode 2: bias+relu, write f32 to global with arbitrary row/col strides
// ---------------------------------------------------------------------------
__device__ __forceinline__ void mlp_layer(
    int mode, const _Float16* __restrict__ X, int rows, int cinPad,
    const _Float16* __restrict__ Wp, const float* __restrict__ bias, int cout,
    _Float16* __restrict__ Y, unsigned* __restrict__ pooledU,
    float* __restrict__ gout, long rowStride, long colStride, long base)
{
  const int tid  = threadIdx.x;
  const int lane = tid & 31;
  const int wv   = tid >> 5;
  const int hib  = lane >> 4;   // which 16-lane half
  const int l15  = lane & 15;
  const int mtiles = rows >> 4;
  const int ntiles = cout >> 4;
  const int ktiles = cinPad >> 5;
  const v16h* WpV = (const v16h*)Wp;

  for (int tile = wv; tile < mtiles * ntiles; tile += (WG / 32)) {
    int mt = tile / ntiles, nt = tile - mt * ntiles;
    int row0 = mt << 4, n0 = nt << 4;

    CFrag acc;
    #pragma unroll
    for (int i = 0; i < 8; ++i) acc.f[i] = 0.f;

    // A fragment rows: per-lane two runs of 4 consecutive dwords -> 2x b128.
    const u32x4* Xr4 = (const u32x4*)(X + (size_t)(row0 + l15) * cinPad);

    // Software-pipelined K loop: fetch kt+1 fragments before kt's WMMA.
    AFrag a0; v16h b0;
    a0.q[0] = Xr4[hib];
    a0.q[1] = Xr4[2 + hib];
    b0 = WpV[(size_t)nt * 32 + lane];

    for (int kt = 0; kt < ktiles; ++kt) {
      AFrag a1; v16h b1;
      int ktn = kt + 1;
      if (ktn < ktiles) {
        a1.q[0] = Xr4[(ktn << 2) + hib];
        a1.q[1] = Xr4[(ktn << 2) + 2 + hib];
        b1 = WpV[((size_t)ktn * ntiles + nt) * 32 + lane];
        if (ktn + 1 < ktiles)
          __builtin_prefetch((const void*)(WpV +
              ((size_t)(ktn + 1) * ntiles + nt) * 32 + lane), 0, 0);
      } else {
        a1 = a0; b1 = b0;
      }
      acc.v = __builtin_amdgcn_wmma_f32_16x16x32_f16(
          false, a0.v, false, b0, (short)0, acc.v, false, false);
      a0 = a1; b0 = b1;
    }

    float bv = bias[n0 + l15];
    if (mode == 0) {
      #pragma unroll
      for (int i = 0; i < 8; ++i) {
        float v = acc.f[i] + bv; v = v > 0.f ? v : 0.f;
        Y[(size_t)(row0 + i + (hib << 3)) * cout + n0 + l15] = (_Float16)v;
      }
    } else if (mode == 1) {
      float pm = 0.f;                                   // relu >= 0
      #pragma unroll
      for (int i = 0; i < 8; ++i) {
        float v = acc.f[i] + bv; v = v > 0.f ? v : 0.f;
        pm = pm > v ? pm : v;
      }
      float o = __shfl_xor(pm, 16, 32);                 // combine rows 0-7 / 8-15
      pm = pm > o ? pm : o;
      if (hib == 0) {
        int g = row0 >> 5;                              // 32-sample group in tile
        atomicMax(&pooledU[g * cout + n0 + l15], __float_as_uint(pm));
      }
    } else {
      #pragma unroll
      for (int i = 0; i < 8; ++i) {
        float v = acc.f[i] + bv; v = v > 0.f ? v : 0.f;
        long row = row0 + i + (hib << 3);
        gout[base + row * rowStride + (long)(n0 + l15) * colStride] = v;
      }
    }
  }
}

// ---------------------------------------------------------------------------
// Weight pre-pack: f32 (cin x cout) -> f16 fragments [kt][nt][lane][16],
// zero-padded to cinPad.
// ---------------------------------------------------------------------------
__global__ void pack_w(const float* __restrict__ W, _Float16* __restrict__ out,
                       int cin, int cinPad, int cout)
{
  int total = (cinPad >> 5) * (cout >> 4) * 512;
  for (int e = blockIdx.x * blockDim.x + threadIdx.x; e < total;
       e += gridDim.x * blockDim.x) {
    int j    = e & 15;
    int lane = (e >> 4) & 31;
    int t    = e >> 9;
    int ntiles = cout >> 4;
    int kt = t / ntiles, nt = t - kt * ntiles;
    int k = (kt << 5) + ((lane >> 4) << 3) + (j & 7) + ((j >> 3) << 4);
    int c = (nt << 4) + (lane & 15);
    float v = (k < cin) ? W[(size_t)k * cout + c] : 0.f;
    out[e] = (_Float16)v;
  }
}

// ---------------------------------------------------------------------------
// Farthest point sampling: one block per batch, iterative argmax reduction.
// Also writes new_xyz (gathered centroids). P = points per thread.
// ---------------------------------------------------------------------------
template<int P>
__global__ __launch_bounds__(WG)
void fps_kernel(const float* __restrict__ xyz, int n, int npoint,
                int* __restrict__ fidx, float* __restrict__ newXyz)
{
  __shared__ float sval[WG];
  __shared__ int   sidx[WG];
  __shared__ int   sfar;
  int b = blockIdx.x, tid = threadIdx.x;
  const float* px = xyz + (size_t)b * n * 3;

  float mind[P];
  #pragma unroll
  for (int j = 0; j < P; ++j) mind[j] = 1e10f;
  int far = 0;

  for (int it = 0; it < npoint; ++it) {
    if (tid == 0) {
      fidx[b * npoint + it] = far;
      newXyz[((size_t)b * npoint + it) * 3 + 0] = px[far * 3 + 0];
      newXyz[((size_t)b * npoint + it) * 3 + 1] = px[far * 3 + 1];
      newXyz[((size_t)b * npoint + it) * 3 + 2] = px[far * 3 + 2];
    }
    float cx = px[far * 3 + 0], cy = px[far * 3 + 1], cz = px[far * 3 + 2];
    float best = -1.f; int bi = 0;
    #pragma unroll
    for (int j = 0; j < P; ++j) {
      int p = tid + (j << 8);
      if (p < n) {
        float dx = px[p * 3 + 0] - cx, dy = px[p * 3 + 1] - cy, dz = px[p * 3 + 2] - cz;
        float d = dx * dx + dy * dy + dz * dz;
        float m = mind[j]; m = m < d ? m : d; mind[j] = m;
        if (m > best) { best = m; bi = p; }
      }
    }
    sval[tid] = best; sidx[tid] = bi;
    __syncthreads();
    for (int s = WG / 2; s > 0; s >>= 1) {
      if (tid < s && sval[tid + s] > sval[tid]) {
        sval[tid] = sval[tid + s]; sidx[tid] = sidx[tid + s];
      }
      __syncthreads();
    }
    if (tid == 0) sfar = sidx[0];
    __syncthreads();
    far = sfar;
  }
}

// ---------------------------------------------------------------------------
// Ball query: one wave per query point; wave32 ballot compaction picks the
// 32 smallest in-radius indices (matches reference top_k semantics).
// ---------------------------------------------------------------------------
__global__ __launch_bounds__(WG)
void ballquery_kernel(const float* __restrict__ xyz, const float* __restrict__ newXyz,
                      int n, int S, float r2, int* __restrict__ outIdx)
{
  int tid = threadIdx.x, lane = tid & 31, wv = tid >> 5;
  int gid = blockIdx.x * (WG / 32) + wv;
  int b = gid / S, s = gid - b * S;
  const float* q = newXyz + ((size_t)b * S + s) * 3;
  float qx = q[0], qy = q[1], qz = q[2];
  const float* px = xyz + (size_t)b * n * 3;
  int* oi = outIdx + ((size_t)b * S + s) * 32;

  int cnt = 0, first = -1;
  for (int base = 0; base < n && cnt < 32; base += 32) {
    int p = base + lane;
    float dx = px[p * 3 + 0] - qx, dy = px[p * 3 + 1] - qy, dz = px[p * 3 + 2] - qz;
    bool in = (dx * dx + dy * dy + dz * dz) < r2;
    unsigned mask = (unsigned)__ballot(in);
    while (mask && cnt < 32) {
      int bit = __ffs(mask) - 1;
      mask &= mask - 1;
      int pidx = base + bit;
      if (first < 0) first = pidx;
      if (lane == 0) oi[cnt] = pidx;
      ++cnt;
    }
  }
  if (first < 0) first = 0;
  if (lane == 0)
    for (int j = cnt; j < 32; ++j) oi[j] = first;
}

// ---------------------------------------------------------------------------
// Fused SA stage: gather (rel-xyz ++ feats) into LDS f16, 3 WMMA MLP layers,
// max-pool over 32 samples. One block = 8 groups = 256 rows.
// ---------------------------------------------------------------------------
__global__ __launch_bounds__(WG)
void sa_kernel(const float* __restrict__ xyz, const float* __restrict__ feats, int Cf,
               const int* __restrict__ ballIdx, const float* __restrict__ newXyz,
               const _Float16* __restrict__ w1, const float* __restrict__ b1,
               const _Float16* __restrict__ w2, const float* __restrict__ b2,
               const _Float16* __restrict__ w3, const float* __restrict__ b3,
               int n, int S, int cinPad, int c1, int c2, int c3,
               float* __restrict__ out)
{
  extern __shared__ char smem[];
  const int ROWS = 256;
  _Float16* X = (_Float16*)smem;                         // ROWS x cinPad
  _Float16* Y = X + (size_t)ROWS * cinPad;               // ROWS x c1
  unsigned* pooledU = (unsigned*)(Y + (size_t)ROWS * c1);// 8 x c3

  int b = blockIdx.y;
  int s0 = blockIdx.x * 8;
  int tid = threadIdx.x;

  { // build input tile: thread == row
    int r = tid, g = r >> 5, sm = r & 31;
    int s = s0 + g;
    int pid = ballIdx[((size_t)b * S + s) * 32 + sm];
    const float* src = xyz + ((size_t)b * n + pid) * 3;
    const float* q   = newXyz + ((size_t)b * S + s) * 3;
    _Float16* xr = X + (size_t)r * cinPad;
    xr[0] = (_Float16)(src[0] - q[0]);
    xr[1] = (_Float16)(src[1] - q[1]);
    xr[2] = (_Float16)(src[2] - q[2]);
    const float* fs = feats + ((size_t)b * n + pid) * Cf;
    for (int c = 0; c < Cf; ++c) xr[3 + c] = (_Float16)fs[c];
    for (int c = 3 + Cf; c < cinPad; ++c) xr[c] = (_Float16)0.f;
  }
  for (int t = tid; t < 8 * c3; t += WG) pooledU[t] = 0u;
  __syncthreads();

  mlp_layer(0, X, ROWS, cinPad, w1, b1, c1, Y, nullptr, nullptr, 0, 0, 0);
  __syncthreads();
  mlp_layer(0, Y, ROWS, c1, w2, b2, c2, X, nullptr, nullptr, 0, 0, 0);
  __syncthreads();
  mlp_layer(1, X, ROWS, c2, w3, b3, c3, nullptr, pooledU, nullptr, 0, 0, 0);
  __syncthreads();

  for (int t = tid; t < 8 * c3; t += WG) {
    int g = t / c3, c = t - g * c3;
    out[((size_t)b * S + s0 + g) * c3 + c] = __uint_as_float(pooledU[t]);
  }
}

// ---------------------------------------------------------------------------
// Fused FP stage: 3-NN interpolation + skip concat into LDS f16, then 2-3
// WMMA MLP layers; final layer writes f32 global (optionally transposed
// straight into d_out). One block = 128 points.
// ---------------------------------------------------------------------------
__global__ __launch_bounds__(WG)
void fp_kernel(const float* __restrict__ uxyz, const float* __restrict__ kxyz,
               const float* __restrict__ skip, int Cs,
               const float* __restrict__ kfeat, int Ck,
               const _Float16* __restrict__ w1, const float* __restrict__ b1,
               const _Float16* __restrict__ w2, const float* __restrict__ b2,
               const _Float16* __restrict__ w3, const float* __restrict__ b3,
               int nu, int nk, int cinPad, int c1, int c2, int c3,
               float* __restrict__ gout, long rowStride, long colStride, long baseMul)
{
  extern __shared__ char smem[];
  const int ROWS = 128;
  _Float16* X = (_Float16*)smem;                // ROWS x cinPad
  _Float16* Y = X + (size_t)ROWS * cinPad;      // ROWS x c1

  int b = blockIdx.y;
  int i0 = blockIdx.x * ROWS;
  int tid = threadIdx.x;

  if (tid < ROWS) {
    int i = i0 + tid;
    const float* q = uxyz + ((size_t)b * nu + i) * 3;
    float qx = q[0], qy = q[1], qz = q[2];
    float d0 = 1e30f, d1 = 1e30f, d2 = 1e30f;
    int j0 = 0, j1 = 0, j2 = 0;
    const float* kp = kxyz + (size_t)b * nk * 3;
    for (int j = 0; j < nk; ++j) {
      float dx = kp[j * 3 + 0] - qx, dy = kp[j * 3 + 1] - qy, dz = kp[j * 3 + 2] - qz;
      float d = dx * dx + dy * dy + dz * dz;
      if (d < d0)      { d2 = d1; j2 = j1; d1 = d0; j1 = j0; d0 = d; j0 = j; }
      else if (d < d1) { d2 = d1; j2 = j1; d1 = d;  j1 = j; }
      else if (d < d2) { d2 = d;  j2 = j; }
    }
    float iw0 = 1.f / (d0 + 1e-8f), iw1 = 1.f / (d1 + 1e-8f), iw2 = 1.f / (d2 + 1e-8f);
    float ws = iw0 + iw1 + iw2; iw0 /= ws; iw1 /= ws; iw2 /= ws;
    const float* f0 = kfeat + ((size_t)b * nk + j0) * Ck;
    const float* f1 = kfeat + ((size_t)b * nk + j1) * Ck;
    const float* f2 = kfeat + ((size_t)b * nk + j2) * Ck;
    _Float16* xr = X + (size_t)tid * cinPad;
    for (int c = 0; c < Ck; ++c)
      xr[c] = (_Float16)(iw0 * f0[c] + iw1 * f1[c] + iw2 * f2[c]);
    const float* sp = skip + ((size_t)b * nu + i) * Cs;
    for (int c = 0; c < Cs; ++c) xr[Ck + c] = (_Float16)sp[c];
    for (int c = Ck + Cs; c < cinPad; ++c) xr[c] = (_Float16)0.f;
  }
  __syncthreads();

  long base = (long)b * baseMul + (long)i0 * rowStride;
  if (c3 == 0) {
    mlp_layer(0, X, ROWS, cinPad, w1, b1, c1, Y, nullptr, nullptr, 0, 0, 0);
    __syncthreads();
    mlp_layer(2, Y, ROWS, c1, w2, b2, c2, nullptr, nullptr, gout, rowStride, colStride, base);
  } else {
    mlp_layer(0, X, ROWS, cinPad, w1, b1, c1, Y, nullptr, nullptr, 0, 0, 0);
    __syncthreads();
    mlp_layer(0, Y, ROWS, c1, w2, b2, c2, X, nullptr, nullptr, 0, 0, 0);
    __syncthreads();
    mlp_layer(2, X, ROWS, c2, w3, b3, c3, nullptr, nullptr, gout, rowStride, colStride, base);
  }
}

// ---------------------------------------------------------------------------
// Host orchestration
// ---------------------------------------------------------------------------
extern "C" void kernel_launch(void* const* d_in, const int* in_sizes, int n_in,
                              void* d_out, int out_size, void* d_ws, size_t ws_size,
                              hipStream_t stream)
{
  (void)in_sizes; (void)n_in; (void)out_size; (void)ws_size;
  const int B = 8, N = 8192;

  const float* pc = (const float*)d_in[0];
  const float *saw[4][3], *sab[4][3], *fpw[4][3] = {}, *fpb[4][3] = {};
  int p = 1;
  for (int i = 0; i < 4; ++i)
    for (int l = 0; l < 3; ++l) { saw[i][l] = (const float*)d_in[p++]; sab[i][l] = (const float*)d_in[p++]; }
  const int fpl[4] = {3, 2, 2, 2};
  for (int i = 0; i < 4; ++i)
    for (int l = 0; l < fpl[i]; ++l) { fpw[i][l] = (const float*)d_in[p++]; fpb[i][l] = (const float*)d_in[p++]; }

  const int   saS[4]      = {2048, 512, 128, 32};
  const float saR[4]      = {0.1f, 0.2f, 0.4f, 0.8f};
  const int   saCf[4]     = {0, 64, 128, 256};
  const int   saCinPad[4] = {32, 96, 160, 288};
  const int   saC[4][3]   = {{32,32,64},{64,64,128},{128,128,256},{256,256,512}};

  const int fpCin[4] = {128, 320, 384, 768};
  const int fpCk[4]  = {128, 256, 256, 512};
  const int fpCs[4]  = {0, 64, 128, 256};
  const int fpC1[4]  = {128, 256, 256, 256};
  const int fpC2[4]  = {128, 128, 256, 256};
  const int fpC3[4]  = {128, 0, 0, 0};

  char* wsp = (char*)d_ws;
  auto carve = [&](size_t bytes) -> void* {
    void* r = (void*)wsp; wsp += (bytes + 255) & ~(size_t)255; return r;
  };

  float* nx[4]; int* fidx[4]; int* bidx[4]; float* saf[4];
  for (int i = 0; i < 4; ++i) nx[i]   = (float*)carve((size_t)B * saS[i] * 3 * 4);
  for (int i = 0; i < 4; ++i) fidx[i] = (int*)carve((size_t)B * saS[i] * 4);
  for (int i = 0; i < 4; ++i) bidx[i] = (int*)carve((size_t)B * saS[i] * 32 * 4);
  for (int i = 0; i < 4; ++i) saf[i]  = (float*)carve((size_t)B * saS[i] * saC[i][2] * 4);
  float* fpf[4] = {nullptr, nullptr, nullptr, nullptr};
  fpf[3] = (float*)carve((size_t)B * 128  * 256 * 4);
  fpf[2] = (float*)carve((size_t)B * 512  * 256 * 4);
  fpf[1] = (float*)carve((size_t)B * 2048 * 128 * 4);

  auto pkElems = [](int cinPad, int cout) -> size_t {
    return (size_t)(cinPad / 32) * (cout / 16) * 512;
  };
  _Float16* saPk[4][3]; _Float16* fpPk[4][3] = {};
  for (int i = 0; i < 4; ++i) {
    int cins[3]    = {3 + saCf[i], saC[i][0], saC[i][1]};
    int cinpads[3] = {saCinPad[i], saC[i][0], saC[i][1]};
    for (int l = 0; l < 3; ++l) {
      size_t e = pkElems(cinpads[l], saC[i][l]);
      saPk[i][l] = (_Float16*)carve(e * 2);
      pack_w<<<dim3((unsigned)((e + 255) / 256)), 256, 0, stream>>>(
          saw[i][l], saPk[i][l], cins[l], cinpads[l], saC[i][l]);
    }
  }
  for (int i = 0; i < 4; ++i) {
    int couts[3] = {fpC1[i], fpC2[i], fpC3[i]};
    int cins[3]  = {fpCin[i], fpC1[i], fpC2[i]};
    for (int l = 0; l < fpl[i]; ++l) {
      size_t e = pkElems(cins[l], couts[l]);
      fpPk[i][l] = (_Float16*)carve(e * 2);
      pack_w<<<dim3((unsigned)((e + 255) / 256)), 256, 0, stream>>>(
          fpw[i][l], fpPk[i][l], cins[l], cins[l], couts[l]);
    }
  }

  (void)hipFuncSetAttribute((const void*)sa_kernel,
      hipFuncAttributeMaxDynamicSharedMemorySize, 300 * 1024);
  (void)hipFuncSetAttribute((const void*)fp_kernel,
      hipFuncAttributeMaxDynamicSharedMemorySize, 300 * 1024);

  // ---- SA chain ----
  const float* curXyz = pc; const float* curFeat = nullptr; int curN = N;
  for (int i = 0; i < 4; ++i) {
    int S = saS[i];
    switch (i) {
      case 0: fps_kernel<32><<<dim3(B), WG, 0, stream>>>(curXyz, curN, S, fidx[i], nx[i]); break;
      case 1: fps_kernel< 8><<<dim3(B), WG, 0, stream>>>(curXyz, curN, S, fidx[i], nx[i]); break;
      case 2: fps_kernel< 2><<<dim3(B), WG, 0, stream>>>(curXyz, curN, S, fidx[i], nx[i]); break;
      default:fps_kernel< 1><<<dim3(B), WG, 0, stream>>>(curXyz, curN, S, fidx[i], nx[i]); break;
    }
    ballquery_kernel<<<dim3((unsigned)(B * S / 8)), WG, 0, stream>>>(
        curXyz, nx[i], curN, S, saR[i] * saR[i], bidx[i]);
    size_t lds = (size_t)256 * saCinPad[i] * 2 + (size_t)256 * saC[i][0] * 2
               + (size_t)8 * saC[i][2] * 4;
    sa_kernel<<<dim3((unsigned)(S / 8), B), WG, lds, stream>>>(
        curXyz, curFeat, saCf[i], bidx[i], nx[i],
        saPk[i][0], sab[i][0], saPk[i][1], sab[i][1], saPk[i][2], sab[i][2],
        curN, S, saCinPad[i], saC[i][0], saC[i][1], saC[i][2], saf[i]);
    curXyz = nx[i]; curFeat = saf[i]; curN = S;
  }

  // ---- FP chain (i = 3,2,1,0) ----
  for (int i = 3; i >= 0; --i) {
    int nu = (i == 0) ? N : saS[i - 1];
    int nk = saS[i];
    const float* uxyz = (i == 0) ? pc : nx[i - 1];
    const float* kxyz = nx[i];
    const float* skip = (i == 0) ? nullptr : saf[i - 1];
    const float* kft  = (i == 3) ? saf[3] : fpf[i + 1];
    size_t lds = (size_t)128 * fpCin[i] * 2 + (size_t)128 * fpC1[i] * 2;
    float* out; long rowStride, colStride, baseMul;
    if (i == 0) {  // final layer writes transposed straight into d_out (B,128,N)
      out = (float*)d_out; rowStride = 1; colStride = N; baseMul = (long)128 * N;
    } else {
      int cout = fpC2[i];
      out = fpf[i]; rowStride = cout; colStride = 1; baseMul = (long)nu * cout;
    }
    fp_kernel<<<dim3((unsigned)(nu / 128), B), WG, lds, stream>>>(
        uxyz, kxyz, skip, fpCs[i], kft, fpCk[i],
        fpPk[i][0], fpb[i][0], fpPk[i][1], fpb[i][1],
        (fpl[i] == 3) ? fpPk[i][2] : nullptr, (fpl[i] == 3) ? fpb[i][2] : nullptr,
        nu, nk, fpCin[i], fpC1[i], fpC2[i], fpC3[i],
        out, rowStride, colStride, baseMul);
  }
}
